// OptimalNeighborhoodSelector_51470888075609
// MI455X (gfx1250) — matrix operand order repaired and checked by reading, not compiled, not run on hardware
//
#include <hip/hip_runtime.h>
#include <hip/hip_fp16.h>

typedef __attribute__((ext_vector_type(16))) _Float16 v16h;
typedef __attribute__((ext_vector_type(8)))  _Float16 v8h;
typedef __attribute__((ext_vector_type(8)))  float    v8f;

#define NPTS   4096
#define CFEAT  64
#define KCAND  64
#define KFINAL 16
#define KPAD   96      // 67 input dims padded to 3 K-chunks of 32
#define HID    128

// ---------------------------------------------------------------------------
// Kernel 0: W1 (67x128 f32) -> padded f16 (96x128), zero rows 67..95
// ---------------------------------------------------------------------------
__global__ void prep_w1_kernel(const float* __restrict__ W1,
                               _Float16* __restrict__ W1h) {
    int i = blockIdx.x * 256 + threadIdx.x;
    if (i < KPAD * HID) {
        int k = i / HID;
        W1h[i] = (k < 67) ? (_Float16)W1[i] : (_Float16)0.0f;
    }
}

// ---------------------------------------------------------------------------
// Kernel 1: exact 64-NN per query point via register d2 + LDS radix-select.
// One 256-thread workgroup per query. Batch xyz staged in LDS (L2->LDS once).
// ---------------------------------------------------------------------------
__global__ __launch_bounds__(256) void knn_kernel(const float* __restrict__ xyz,
                                                  int* __restrict__ cand) {
    __shared__ float sx[NPTS], sy[NPTS], sz[NPTS];          // 48KB
    __shared__ unsigned hist[256];
    __shared__ unsigned scan[256];
    __shared__ unsigned sPrefix, sK, sLT, sEQ;

    const int q   = blockIdx.x;
    const int b   = q >> 12;              // q / 4096
    const int n   = q & (NPTS - 1);
    const int tid = threadIdx.x;

    const float* bx = xyz + (size_t)b * NPTS * 3;
    for (int i = tid; i < NPTS; i += 256) {
        sx[i] = bx[3 * i + 0];
        sy[i] = bx[3 * i + 1];
        sz[i] = bx[3 * i + 2];
    }
    if (tid == 0) { sPrefix = 0u; sK = KCAND; sLT = 0u; sEQ = 0u; }
    __syncthreads();

    const float qx = sx[n], qy = sy[n], qz = sz[n];

    // 16 distance keys per thread, monotonic uint encoding of d2
    unsigned key[16];
#pragma unroll
    for (int i = 0; i < 16; ++i) {
        int j = tid + (i << 8);
        float dx = sx[j] - qx, dy = sy[j] - qy, dz = sz[j] - qz;
        float d2 = dx * dx + dy * dy + dz * dz;
        unsigned u = __float_as_uint(d2);
        u ^= (u >> 31) ? 0xFFFFFFFFu : 0x80000000u;   // order-preserving
        if (j == n) u = 0xFFFFFFFFu;                  // exclude self (ref drops nearest)
        key[i] = u;
    }

    // 4-pass MSB radix select: find T = 64th smallest key exactly
    unsigned prefmask = 0u;
    for (int pass = 0; pass < 4; ++pass) {
        const int shift = 24 - 8 * pass;
        hist[tid] = 0u;
        __syncthreads();
        const unsigned pfx = sPrefix;
#pragma unroll
        for (int i = 0; i < 16; ++i) {
            unsigned k = key[i];
            if ((k & prefmask) == pfx)
                atomicAdd(&hist[(k >> shift) & 0xFFu], 1u);
        }
        __syncthreads();
        scan[tid] = hist[tid];
        __syncthreads();
        for (int off = 1; off < 256; off <<= 1) {
            unsigned add = (tid >= off) ? scan[tid - off] : 0u;
            __syncthreads();
            scan[tid] += add;
            __syncthreads();
        }
        const unsigned incl = scan[tid];
        const unsigned excl = incl - hist[tid];
        const unsigned kk   = sK;
        __syncthreads();                 // everyone has read sK/sPrefix
        if (excl < kk && kk <= incl) {   // exactly one thread matches
            sPrefix = pfx | ((unsigned)tid << shift);
            sK      = kk - excl;
        }
        prefmask |= (0xFFu << shift);
        __syncthreads();
    }

    const unsigned T      = sPrefix;
    const unsigned needEq = sK;                 // #elements == T to take
    const unsigned nLT    = KCAND - needEq;     // #elements  < T
    int* oq = cand + (size_t)q * KCAND;

#pragma unroll
    for (int i = 0; i < 16; ++i) {
        unsigned k = key[i];
        int j = tid + (i << 8);
        if (k < T) {
            unsigned p = atomicAdd(&sLT, 1u);
            oq[p] = j;
        } else if (k == T) {
            unsigned p = atomicAdd(&sEQ, 1u);
            if (p < needEq) oq[nLT + p] = j;
        }
    }
}

// ---------------------------------------------------------------------------
// Kernel 2: per-point 64x96 @ 96x128 f16 WMMA MLP + GELU + W2 dot + top-16.
// 128 threads = 4 waves; wave w owns candidate rows [16w, 16w+16).
// ---------------------------------------------------------------------------
__global__ __launch_bounds__(128) void mlp_topk_kernel(
    const float* __restrict__ xyz, const float* __restrict__ feat,
    const _Float16* __restrict__ W1h, const float* __restrict__ b1,
    const float* __restrict__ W2, const float* __restrict__ b2,
    const int* __restrict__ cand, int* __restrict__ out)
{
    __shared__ _Float16 sA[KCAND * KPAD];   // 12KB: A tiles, row-major [m][k]
    __shared__ _Float16 sB[KPAD * HID];     // 24KB: W1 f16, row-major [k][n]
    __shared__ float sScore[KCAND];

    const int q   = blockIdx.x;
    const int b   = q >> 12;
    const int n   = q & (NPTS - 1);
    const int tid = threadIdx.x;

    // Stage W1h into LDS (L2-hot, 24KB, 12 x uint4 per thread)
    {
        const uint4* src = (const uint4*)W1h;
        uint4* dst = (uint4*)sB;
        for (int i = tid; i < (KPAD * HID) / 8; i += 128) dst[i] = src[i];
    }

    // Build A: two threads per candidate row; k0..2 xyz diff, k3..66 feat diff
    const float* bxyz  = xyz  + (size_t)b * NPTS * 3;
    const float* bfeat = feat + (size_t)b * NPTS * CFEAT;
    const int* cq = cand + (size_t)q * KCAND;
    {
        const int r  = tid >> 1;
        const int hh = tid & 1;
        const int c  = cq[r];
        _Float16* row = sA + r * KPAD;
        const float4* fc = (const float4*)(bfeat + (size_t)c * CFEAT);
        const float4* fq = (const float4*)(bfeat + (size_t)n * CFEAT);
        if (hh == 0) {
            row[0] = (_Float16)(bxyz[c * 3 + 0] - bxyz[n * 3 + 0]);
            row[1] = (_Float16)(bxyz[c * 3 + 1] - bxyz[n * 3 + 1]);
            row[2] = (_Float16)(bxyz[c * 3 + 2] - bxyz[n * 3 + 2]);
#pragma unroll
            for (int v = 0; v < 8; ++v) {
                float4 a = fc[v], p = fq[v];
                row[3 + 4 * v + 0] = (_Float16)(a.x - p.x);
                row[3 + 4 * v + 1] = (_Float16)(a.y - p.y);
                row[3 + 4 * v + 2] = (_Float16)(a.z - p.z);
                row[3 + 4 * v + 3] = (_Float16)(a.w - p.w);
            }
        } else {
#pragma unroll
            for (int v = 8; v < 16; ++v) {
                float4 a = fc[v], p = fq[v];
                row[3 + 4 * v + 0] = (_Float16)(a.x - p.x);
                row[3 + 4 * v + 1] = (_Float16)(a.y - p.y);
                row[3 + 4 * v + 2] = (_Float16)(a.z - p.z);
                row[3 + 4 * v + 3] = (_Float16)(a.w - p.w);
            }
#pragma unroll
            for (int k = 67; k < KPAD; ++k) row[k] = (_Float16)0.0f;
        }
    }
    __syncthreads();

    const int wave  = tid >> 5;
    const int lane  = tid & 31;
    const int lmod  = lane & 15;
    const int lhalf = lane >> 4;
    const int m0    = wave * 16;

    // A fragments: lane = M row; elems 0..7 -> K = 8*lhalf + j,
    //              elems 8..15 -> K = 16 + 8*lhalf + (j-8)   (ISA 16-bit A layout)
    union U16 { v16h v; v8h h[2]; };
    v16h afrag[3];
#pragma unroll
    for (int kc = 0; kc < 3; ++kc) {
        U16 u;
        const _Float16* base = sA + (m0 + lmod) * KPAD + kc * 32 + lhalf * 8;
        u.h[0] = *(const v8h*)(base);        // K = kc*32 + 8*lhalf + 0..7
        u.h[1] = *(const v8h*)(base + 16);   // K = kc*32 + 16 + 8*lhalf + 0..7
        afrag[kc] = u.v;
    }

    v8f acc[8];
#pragma unroll
    for (int nt = 0; nt < 8; ++nt) {
        v8f z = {0.f, 0.f, 0.f, 0.f, 0.f, 0.f, 0.f, 0.f};
        acc[nt] = z;
    }

    // B fragments: lane column N = nt*16 + lmod; elems 0..7 -> K = kc*32 + 8*lhalf + j,
    //              elems 8..15 -> K = kc*32 + 16 + 8*lhalf + (j-8)
#pragma unroll
    for (int kc = 0; kc < 3; ++kc) {
        const int kb = kc * 32 + lhalf * 8;
        const _Float16* bcol = sB + kb * HID + lmod;
#pragma unroll
        for (int nt = 0; nt < 8; ++nt) {
            v16h bfrag;
#pragma unroll
            for (int s = 0; s < 8; ++s) {
                bfrag[s]     = bcol[s * HID + nt * 16];
                bfrag[8 + s] = bcol[(16 + s) * HID + nt * 16];
            }
            acc[nt] = __builtin_amdgcn_wmma_f32_16x16x32_f16(
                false, afrag[kc], false, bfrag, (short)0, acc[nt], false, false);
        }
    }

    // C layout: elem r -> M = m0 + r + 8*lhalf; lane -> N = nt*16 + lmod.
    // Fuse bias + exact GELU + W2 dot, then reduce over the 16 N-lanes.
    float partial[8];
#pragma unroll
    for (int r = 0; r < 8; ++r) partial[r] = 0.f;
#pragma unroll
    for (int nt = 0; nt < 8; ++nt) {
        const int nn = nt * 16 + lmod;
        const float w2v = W2[nn];
        const float b1v = b1[nn];
#pragma unroll
        for (int r = 0; r < 8; ++r) {
            float h = acc[nt][r] + b1v;
            float g = 0.5f * h * (1.0f + erff(h * 0.70710678118654752f));
            partial[r] += g * w2v;
        }
    }
#pragma unroll
    for (int off = 1; off < 16; off <<= 1) {
#pragma unroll
        for (int r = 0; r < 8; ++r)
            partial[r] += __shfl_xor(partial[r], off, 32);
    }
    if (lmod == 0) {
        const float bias2 = b2[0];
#pragma unroll
        for (int r = 0; r < 8; ++r)
            sScore[m0 + lhalf * 8 + r] = partial[r] + bias2;
    }
    __syncthreads();

    // Wave 0: iterative top-16 (descending score, smaller slot wins ties)
    if (wave == 0) {
        int* oq = out + (size_t)q * KFINAL;
        float s0 = sScore[lane], s1 = sScore[lane + 32];
        const int c0 = cq[lane], c1 = cq[lane + 32];
        bool alive0 = true, alive1 = true;
        for (int it = 0; it < KFINAL; ++it) {
            unsigned u0 = __float_as_uint(s0);
            u0 ^= (u0 >> 31) ? 0xFFFFFFFFu : 0x80000000u;
            unsigned u1 = __float_as_uint(s1);
            u1 ^= (u1 >> 31) ? 0xFFFFFFFFu : 0x80000000u;
            unsigned long long k0 =
                alive0 ? (((unsigned long long)u0 << 8) | (unsigned)(63 - lane)) : 0ull;
            unsigned long long k1 =
                alive1 ? (((unsigned long long)u1 << 8) | (unsigned)(31 - lane)) : 0ull;
            unsigned long long best = (k0 > k1) ? k0 : k1;
#pragma unroll
            for (int off = 1; off < 32; off <<= 1) {
                unsigned long long o = __shfl_xor(best, off, 32);
                if (o > best) best = o;
            }
            const int slot = 63 - (int)(best & 0xFFull);
            if (slot == lane)           { oq[it] = c0; alive0 = false; }
            else if (slot == lane + 32) { oq[it] = c1; alive1 = false; }
        }
    }
}

// ---------------------------------------------------------------------------
// Launch: inputs = {xyz, features, W1, b1, W2, b2}; output = int32 (B,N,16)
// Workspace: [cand_idx: B*N*64 ints][W1h: 96*128 f16]
// ---------------------------------------------------------------------------
extern "C" void kernel_launch(void* const* d_in, const int* in_sizes, int n_in,
                              void* d_out, int out_size, void* d_ws, size_t ws_size,
                              hipStream_t stream) {
    const float* xyz  = (const float*)d_in[0];
    const float* feat = (const float*)d_in[1];
    const float* W1   = (const float*)d_in[2];
    const float* b1   = (const float*)d_in[3];
    const float* W2   = (const float*)d_in[4];
    const float* b2   = (const float*)d_in[5];
    int* out = (int*)d_out;

    const int BN = in_sizes[0] / 3;   // B * 4096

    int* cand = (int*)d_ws;
    _Float16* W1h = (_Float16*)((char*)d_ws + (size_t)BN * KCAND * sizeof(int));

    prep_w1_kernel<<<(KPAD * HID + 255) / 256, 256, 0, stream>>>(W1, W1h);
    knn_kernel<<<BN, 256, 0, stream>>>(xyz, cand);
    mlp_topk_kernel<<<BN, 128, 0, stream>>>(xyz, feat, W1h, b1, W2, b2, cand, out);
}
